// HRNN_87067577024944
// MI455X (gfx1250) — matrix operand order repaired and checked by reading, not compiled, not run on hardware
//
#include <hip/hip_runtime.h>
#include <hip/hip_bf16.h>
#include <math.h>

// ---------------------------------------------------------------------------
// HRNN for MI455X (gfx1250): persistent wave32 WMMA-bf16 recurrent kernel.
//   T=64, B=32, DIN=32, E=8, NB=4, DM=512, DH=2048, DOUT=512
// Pass 1: transpose+convert all fp32 weights -> bf16 [out][in] in workspace
//         (147 MB bf16 -> fits the 192 MB L2, re-streamed 64x at L2 speed).
// Pass 2: 8 persistent workgroups (one per encoder), full state in 194KB LDS,
//         every matmul via v_wmma_f32_16x16x32_bf16, f32 accumulate,
//         activations via raw v_exp/v_rcp transcendentals (branch-free).
// ---------------------------------------------------------------------------

typedef unsigned short u16;
typedef __bf16 bf16_t;
typedef bf16_t v16bf __attribute__((ext_vector_type(16)));
typedef float  v8f   __attribute__((ext_vector_type(8)));
typedef u16    su8   __attribute__((ext_vector_type(8)));

union BF16x16 { su8 h2[2]; v16bf v; };

#define T_STEPS 64
#define NE      8
#define NBLK    4
#define DMID    512
#define DHID    2048
#define DOUTV   512
#define BLOCK   512          // 16 waves of 32
#define NWAVES  16
#define NTPW    2            // n-tiles (of 16 cols) per wave: 32 tiles / 16 waves

__device__ __forceinline__ u16 f32_to_bf16(float f) {
  unsigned u = __float_as_uint(f);
  u += 0x7FFFu + ((u >> 16) & 1u);           // round-to-nearest-even
  return (u16)(u >> 16);
}

// Branch-free tanh on the transcendental pipe:
//   tanh(x) = 1 - 2/(exp2(2*log2e*x) + 1); saturates correctly at +/-inf.
__device__ __forceinline__ float fast_tanh(float x) {
#if __has_builtin(__builtin_amdgcn_exp2f) && __has_builtin(__builtin_amdgcn_rcpf)
  float e = __builtin_amdgcn_exp2f(x * 2.885390081777927f);   // 2/ln(2)... = 2*log2(e)
  return 1.0f - 2.0f * __builtin_amdgcn_rcpf(e + 1.0f);
#else
  return tanhf(x);
#endif
}

// tanh-form GELU (error ~1e-3, below bf16 input quantization)
__device__ __forceinline__ float fast_gelu(float x) {
  float u = 0.7978845608028654f * x * (1.0f + 0.044715f * x * x);
  return 0.5f * x * (1.0f + fast_tanh(u));
}

// A-tile 16x32 bf16 from row-major LDS activation [rows][ldk].
// ISA layout: lane<16: K= k0+{0..7}, k0+16+{0..7}; lane>=16: +8 on both halves.
__device__ __forceinline__ v16bf load_tile_A(const u16* p, int ldk, int m0, int k0, int lane) {
  int row = m0 + (lane & 15);
  int khi = (lane >> 4) & 1;
  const u16* q = p + row * ldk + k0 + khi * 8;
  BF16x16 r;
  r.h2[0] = *(const su8*)(q);
  r.h2[1] = *(const su8*)(q + 16);
  return r.v;
}

// B-tile 32x16 bf16 from transposed weights Wt[out][in] (row-major, ldk = in).
// ISA layout: N = lane%16, K = k0+{0..7}, k0+16+{0..7}; lane>=16: +8 on both halves.
__device__ __forceinline__ v16bf load_tile_B(const u16* __restrict__ w, int ldk,
                                             int n0, int k0, int lane) {
  int row = n0 + (lane & 15);
  int khi = (lane >> 4) & 1;
  const u16* q = w + (size_t)row * ldk + k0 + khi * 8;
  BF16x16 r;
  r.h2[0] = *(const su8*)(q);
  r.h2[1] = *(const su8*)(q + 16);
  return r.v;
}

__device__ __forceinline__ v8f wmma_bf16(v16bf a, v16bf b, v8f c) {
  return __builtin_amdgcn_wmma_f32_16x16x32_bf16(false, a, false, b, (short)0, c, false, false);
}

// D-tile (16x16 f32): apply bias + activation, write bf16 into LDS.
// C/D layout: N = lane%16, M = vgpr + 8*(lane>=16).
__device__ __forceinline__ void store_tile_act(u16* outp, int ldk, int m0, int n0,
                                               v8f acc, const float* __restrict__ bias,
                                               int act, int lane) {
  int n  = n0 + (lane & 15);
  int mb = m0 + ((lane >> 4) & 1) * 8;
  float bv = bias[n];
#pragma unroll
  for (int v = 0; v < 8; ++v) {
    float xv = acc[v] + bv;
    float y = (act == 0) ? fast_tanh(xv) : fast_gelu(xv);
    outp[(mb + v) * ldk + n] = f32_to_bf16(y);
  }
}

// One 512-wide layer: OUT[32,512] = act( A0*W[0:k0] + A1*W[k0:k0+k1] + b ).
// All 16 waves cooperate; wave owns NTPW n-tiles x 2 m-tiles. No syncs inside.
__device__ __forceinline__ void layer512(
    const u16* a0p, int a0ldk, int a0klen,
    const u16* a1p, int a1ldk, int a1klen,
    const u16* __restrict__ wt, int ktot,
    const float* __restrict__ bias,
    u16* outp, int oldk, int act, int wave, int lane)
{
  const v8f vzero = {0.f,0.f,0.f,0.f,0.f,0.f,0.f,0.f};
  v8f acc[2][NTPW];
#pragma unroll
  for (int m = 0; m < 2; ++m)
#pragma unroll
    for (int j = 0; j < NTPW; ++j) acc[m][j] = vzero;

  int kglob = 0;
#pragma unroll 1
  for (int seg = 0; seg < 2; ++seg) {
    const u16* ap = seg ? a1p : a0p;
    int aldk = seg ? a1ldk : a0ldk;
    int klen = seg ? a1klen : a0klen;
    for (int k = 0; k < klen; k += 32) {
      v16bf a0 = load_tile_A(ap, aldk, 0,  k, lane);
      v16bf a1 = load_tile_A(ap, aldk, 16, k, lane);
#pragma unroll
      for (int j = 0; j < NTPW; ++j) {
        int n0 = (wave * NTPW + j) << 4;
        v16bf b = load_tile_B(wt, ktot, n0, kglob, lane);
        acc[0][j] = wmma_bf16(a0, b, acc[0][j]);
        acc[1][j] = wmma_bf16(a1, b, acc[1][j]);
      }
      // pull next K-slab of this wave's weights toward the WGP (L2-resident)
      __builtin_prefetch(wt + (size_t)((wave * NTPW) << 4) * ktot + kglob + 32, 0, 1);
      kglob += 32;
    }
  }
#pragma unroll
  for (int j = 0; j < NTPW; ++j) {
    int n0 = (wave * NTPW + j) << 4;
    store_tile_act(outp, oldk, 0,  n0, acc[0][j], bias, act, lane);
    store_tile_act(outp, oldk, 16, n0, acc[1][j], bias, act, lane);
  }
}

// fp32 [nmat][rows][cols] -> bf16 [nmat][cols][rows] (weights pre-transpose)
__global__ void transpose_bf16(const float* __restrict__ src, u16* __restrict__ dst,
                               int nmat, int rows, int cols) {
  size_t per = (size_t)rows * cols;
  size_t total = (size_t)nmat * per;
  size_t stride = (size_t)gridDim.x * blockDim.x;
  for (size_t i = (size_t)blockIdx.x * blockDim.x + threadIdx.x; i < total; i += stride) {
    size_t m = i / per, r = i % per;
    int ri = (int)(r / cols), co = (int)(r % cols);
    dst[m * per + (size_t)co * rows + ri] = f32_to_bf16(src[i]);
  }
}

__global__ void hrnn_persistent(
    const float* __restrict__ x,      const float* __restrict__ h0,
    const u16* __restrict__ Wb0l0t,   const float* __restrict__ bb0l0,
    const u16* __restrict__ Wb0l1t,   const float* __restrict__ bb0l1,
    const u16* __restrict__ Wb0l2t,   const float* __restrict__ bb0l2,
    const u16* __restrict__ Wbl0t,    const float* __restrict__ bbl0,
    const u16* __restrict__ Wbl12t,   const float* __restrict__ bbl12,
    const u16* __restrict__ Wf1t,     const float* __restrict__ bf1,
    const u16* __restrict__ Wf2t,     const float* __restrict__ bf2,
    float* __restrict__ out)
{
  extern __shared__ u16 lds[];
  u16* hbuf = lds;                        // [32][2048] recurrent state (bf16) 128KB
  u16* xbuf = hbuf + 32 * DHID;           // [32][32]   x_t staging            2KB
  u16* zA   = xbuf + 32 * 32;             // [32][512]  ping                  32KB
  u16* zB   = zA   + 32 * DMID;           // [32][512]  pong                  32KB

  const int e    = blockIdx.x;
  const int tid  = threadIdx.x;
  const int wave = tid >> 5;
  const int lane = tid & 31;
  const v8f vzero = {0.f,0.f,0.f,0.f,0.f,0.f,0.f,0.f};

  // h <- h0[e]  ([E,NB,B,DM] fp32 -> [B][NB*DM] bf16)
  for (int i = tid; i < 32 * DHID; i += BLOCK) {
    int b = i >> 11, c = i & 2047, kb = c >> 9, o = c & 511;
    hbuf[i] = f32_to_bf16(h0[(((size_t)e * NBLK + kb) * 32 + b) * DMID + o]);
  }
  __syncthreads();

  for (int t = 0; t < T_STEPS; ++t) {
    // stage x_t [B,DIN] (broadcast across encoders)
    for (int i = tid; i < 32 * 32; i += BLOCK)
      xbuf[i] = f32_to_bf16(x[(size_t)t * 32 * 32 + i]);
    __syncthreads();

    // ---------------- block 0: concat(x, h[0]) -> 3x tanh ----------------
    layer512(xbuf, 32, 32, hbuf, DHID, DMID,
             Wb0l0t + (size_t)e * DMID * 544, 544, bb0l0 + e * DMID,
             zA, DMID, 0, wave, lane);
    __syncthreads();
    layer512(zA, DMID, DMID, nullptr, 0, 0,
             Wb0l1t + (size_t)e * DMID * DMID, DMID, bb0l1 + e * DMID,
             zB, DMID, 0, wave, lane);
    __syncthreads();
    layer512(zB, DMID, DMID, nullptr, 0, 0,
             Wb0l2t + (size_t)e * DMID * DMID, DMID, bb0l2 + e * DMID,
             hbuf /*cols 0..511*/, DHID, 0, wave, lane);
    __syncthreads();

    // ---------------- blocks 1..3: concat(prev, h[kb]) -> 3x tanh --------
    for (int kb = 1; kb < NBLK; ++kb) {
      const u16*   w0 = Wbl0t + ((size_t)e * 3 + (kb - 1)) * (DMID * 1024);
      const float* b0 = bbl0  + ((size_t)e * 3 + (kb - 1)) * DMID;
      layer512(hbuf + (kb - 1) * DMID, DHID, DMID,   // prev = new h[kb-1]
               hbuf + kb * DMID,       DHID, DMID,   // old  h[kb]
               w0, 1024, b0, zA, DMID, 0, wave, lane);
      __syncthreads();
      const u16*   w1 = Wbl12t + (((size_t)e * 3 + (kb - 1)) * 2 + 0) * (DMID * DMID);
      const float* b1 = bbl12  + (((size_t)e * 3 + (kb - 1)) * 2 + 0) * DMID;
      layer512(zA, DMID, DMID, nullptr, 0, 0, w1, DMID, b1, zB, DMID, 0, wave, lane);
      __syncthreads();
      layer512(zB, DMID, DMID, nullptr, 0, 0, w1 + DMID * DMID, DMID, b1 + DMID,
               hbuf + kb * DMID, DHID, 0, wave, lane);
      __syncthreads();
    }

    // ---------------- FFN: GELU(h*Wf1+bf1)*Wf2+bf2, chunked 4x512 --------
    v8f oacc[2][NTPW];
#pragma unroll
    for (int m = 0; m < 2; ++m)
#pragma unroll
      for (int j = 0; j < NTPW; ++j) oacc[m][j] = vzero;

    const u16* wf2e = Wf2t + (size_t)e * DOUTV * DHID;
    for (int c = 0; c < 4; ++c) {
      layer512(hbuf, DHID, DHID, nullptr, 0, 0,
               Wf1t + ((size_t)e * DHID + c * DMID) * DHID, DHID,
               bf1 + (size_t)e * DHID + c * DMID,
               zA, DMID, 1 /*GELU*/, wave, lane);
      __syncthreads();
      for (int k = 0; k < DMID; k += 32) {
        v16bf a0 = load_tile_A(zA, DMID, 0,  k, lane);
        v16bf a1 = load_tile_A(zA, DMID, 16, k, lane);
#pragma unroll
        for (int j = 0; j < NTPW; ++j) {
          int n0 = (wave * NTPW + j) << 4;
          v16bf b = load_tile_B(wf2e, DHID, n0, c * DMID + k, lane);
          oacc[0][j] = wmma_bf16(a0, b, oacc[0][j]);
          oacc[1][j] = wmma_bf16(a1, b, oacc[1][j]);
        }
      }
      __syncthreads();   // zA reused next chunk
    }

    // ---------------- write y_t : out[(e*B+b)][t*DOUT+o], fp32 -----------
    {
      int nl = lane & 15;
      int mb = ((lane >> 4) & 1) * 8;
#pragma unroll
      for (int j = 0; j < NTPW; ++j) {
        int n = ((wave * NTPW + j) << 4) + nl;
        float bb = bf2[(size_t)e * DOUTV + n];
#pragma unroll
        for (int mt = 0; mt < 2; ++mt)
#pragma unroll
          for (int v = 0; v < 8; ++v) {
            int brow = mt * 16 + mb + v;
            out[((size_t)(e * 32 + brow) * T_STEPS + t) * DOUTV + n] = oacc[mt][j][v] + bb;
          }
      }
    }
    __syncthreads();
  }
}

extern "C" void kernel_launch(void* const* d_in, const int* in_sizes, int n_in,
                              void* d_out, int out_size, void* d_ws, size_t ws_size,
                              hipStream_t stream) {
  const float* x     = (const float*)d_in[0];
  const float* h0    = (const float*)d_in[1];
  const float* Wb0l0 = (const float*)d_in[2];  const float* bb0l0 = (const float*)d_in[3];
  const float* Wb0l1 = (const float*)d_in[4];  const float* bb0l1 = (const float*)d_in[5];
  const float* Wb0l2 = (const float*)d_in[6];  const float* bb0l2 = (const float*)d_in[7];
  const float* Wbl0  = (const float*)d_in[8];  const float* bbl0  = (const float*)d_in[9];
  const float* Wbl12 = (const float*)d_in[10]; const float* bbl12 = (const float*)d_in[11];
  const float* Wf1   = (const float*)d_in[12]; const float* bf1   = (const float*)d_in[13];
  const float* Wf2   = (const float*)d_in[14]; const float* bf2   = (const float*)d_in[15];
  float* out = (float*)d_out;

  // bf16 transposed-weight workspace layout (147 MB total -> L2-resident)
  u16* p = (u16*)d_ws;
  u16* Wb0l0t = p; p += (size_t)NE * 544 * 512;
  u16* Wb0l1t = p; p += (size_t)NE * 512 * 512;
  u16* Wb0l2t = p; p += (size_t)NE * 512 * 512;
  u16* Wbl0t  = p; p += (size_t)NE * 3 * 1024 * 512;
  u16* Wbl12t = p; p += (size_t)NE * 3 * 2 * 512 * 512;
  u16* Wf1t   = p; p += (size_t)NE * 2048 * 2048;
  u16* Wf2t   = p; p += (size_t)NE * 2048 * 512;

  dim3 tb(256);
  transpose_bf16<<<4096, tb, 0, stream>>>(Wb0l0, Wb0l0t, NE, 544, 512);
  transpose_bf16<<<4096, tb, 0, stream>>>(Wb0l1, Wb0l1t, NE, 512, 512);
  transpose_bf16<<<4096, tb, 0, stream>>>(Wb0l2, Wb0l2t, NE, 512, 512);
  transpose_bf16<<<4096, tb, 0, stream>>>(Wbl0,  Wbl0t,  NE * 3, 1024, 512);
  transpose_bf16<<<4096, tb, 0, stream>>>(Wbl12, Wbl12t, NE * 6, 512, 512);
  transpose_bf16<<<8192, tb, 0, stream>>>(Wf1,   Wf1t,   NE, 2048, 2048);
  transpose_bf16<<<4096, tb, 0, stream>>>(Wf2,   Wf2t,   NE, 2048, 512);

  // 194 KB dynamic LDS/WG (legal on CDNA5: 320 KB/WGP)
  const size_t ldsb = (size_t)(32 * DHID + 32 * 32 + 2 * 32 * DMID) * sizeof(u16);
  (void)hipFuncSetAttribute((const void*)hrnn_persistent,
                            hipFuncAttributeMaxDynamicSharedMemorySize, (int)ldsb);
  hrnn_persistent<<<NE, BLOCK, ldsb, stream>>>(
      x, h0, Wb0l0t, bb0l0, Wb0l1t, bb0l1, Wb0l2t, bb0l2,
      Wbl0t, bbl0, Wbl12t, bbl12, Wf1t, bf1, Wf2t, bf2, out);
}
